// FFNNTransducerModel_1975684956533
// MI455X (gfx1250) — compile-verified, matrix operand order checked by reading, not run on hardware
//
#include <hip/hip_runtime.h>
#include <hip/hip_bf16.h>
#include <stddef.h>
#include <stdint.h>

// ---------------------------------------------------------------------------
// Problem constants (from reference)
// ---------------------------------------------------------------------------
#define V_   128
#define H_   2
#define EMB_ 128
#define ENC_ 512
#define PRED_ 256
#define JOIN_ 512
#define B_   8
#define T_   512
#define U_   64
#define UP1_ 65                       // U+1
#define ROWS_PER_B (T_ * UP1_)        // 33280 = 130 * 256

typedef __attribute__((ext_vector_type(16))) _Float16 v16h;
typedef __attribute__((ext_vector_type(2)))  __fp16   v2h;   // cvt_pkrtz result type
typedef __attribute__((ext_vector_type(8)))  float    v8f;
typedef __attribute__((ext_vector_type(2)))  float    v2f;

union AFrag { v16h v; v2h p[8]; };   // same 32-byte bit layout

static __device__ __forceinline__ v2h pk2(float x, float y) {
  return __builtin_amdgcn_cvt_pkrtz(x, y);   // v_cvt_pk_rtz_f16_f32
}

// ---------------------------------------------------------------------------
// Pack an f32 row-major KxN weight into f16 WMMA B-fragment order.
//   lanes 0-15 : N = nt*16 + lane,  VGPR j in 0..3 -> K = kb+2j,2j+1 (kb = kt*32)
//                                   VGPR j in 4..7 -> K = kb+16+2(j-4), ...
//   lanes 16-31: same N,            K blocks offset by +8
// One fragment = 32 lanes * 32 bytes = 1024 B, stored lane-major so the GEMM
// loads it with one 32B/lane vector load (2x b128).
// ---------------------------------------------------------------------------
__global__ __launch_bounds__(256) void pack_b_kernel(
    const float* __restrict__ W, _Float16* __restrict__ dst,
    int N, int nTiles) {
  int f = blockIdx.x * 8 + (threadIdx.x >> 5);   // fragment id (one per wave)
  int L = threadIdx.x & 31;
  int kt = f / nTiles;
  int nt = f - kt * nTiles;
  int nn = nt * 16 + (L & 15);
  int kb = kt * 32 + ((L >> 4) & 1) * 8;
  AFrag a;
#pragma unroll
  for (int j = 0; j < 4; ++j) {
    int k0 = kb + 2 * j;
    a.p[j]     = pk2(W[(size_t)(k0)      * N + nn], W[(size_t)(k0 + 1)  * N + nn]);
    a.p[4 + j] = pk2(W[(size_t)(k0 + 16) * N + nn], W[(size_t)(k0 + 17) * N + nn]);
  }
  *(v16h*)(dst + (size_t)f * 512 + (size_t)L * 16) = a.v;
}

// ---------------------------------------------------------------------------
// Prediction network: embedding context gather + 2-layer ReLU MLP. Tiny.
// ---------------------------------------------------------------------------
__global__ __launch_bounds__(256) void pred_mlp_kernel(
    const int* __restrict__ targets, const float* __restrict__ emb,
    const float* __restrict__ W1, const float* __restrict__ b1,
    const float* __restrict__ W2, const float* __restrict__ b2,
    float* __restrict__ pred) {
  __shared__ float se[256];
  __shared__ float sp[256];
  int row = blockIdx.x;                 // 0..519
  int b = row / UP1_;
  int u = row - b * UP1_;
  int j = threadIdx.x;
  int tok0 = (u >= 1) ? targets[b * U_ + (u - 1)] : (V_ - 1);
  int tok1 = (u >= 2) ? targets[b * U_ + (u - 2)] : (V_ - 1);
  se[j] = (j < EMB_) ? emb[(size_t)tok0 * EMB_ + j]
                     : emb[(size_t)tok1 * EMB_ + (j - EMB_)];
  __syncthreads();
  float s = b1[j];
  for (int k = 0; k < 2 * EMB_; ++k) s += se[k] * W1[(size_t)k * PRED_ + j];
  s = fmaxf(s, 0.0f);
  __syncthreads();
  sp[j] = s;
  __syncthreads();
  float s2 = b2[j];
  for (int k = 0; k < PRED_; ++k) s2 += sp[k] * W2[(size_t)k * PRED_ + j];
  pred[(size_t)row * PRED_ + j] = fmaxf(s2, 0.0f);
}

// ---------------------------------------------------------------------------
// pred_proj = pred @ Wp + bj1   (Wp = rows ENC..ENC+PRED-1 of Wj1). Tiny.
// ---------------------------------------------------------------------------
__global__ __launch_bounds__(256) void predp_kernel(
    const float* __restrict__ pred, const float* __restrict__ Wj1,
    const float* __restrict__ bj1, float* __restrict__ predp) {
  __shared__ float sp[PRED_];
  int row = blockIdx.x;                               // 0..519
  int col = blockIdx.y * 256 + threadIdx.x;           // 0..511
  sp[threadIdx.x] = pred[(size_t)row * PRED_ + threadIdx.x];
  __syncthreads();
  float s = bj1[col];
  for (int k = 0; k < PRED_; ++k)
    s += sp[k] * Wj1[(size_t)(ENC_ + k) * JOIN_ + col];
  predp[(size_t)row * JOIN_ + col] = s;
}

// ---------------------------------------------------------------------------
// enc_proj = encoder_states @ We  via f16 WMMA, f32 accumulate.
// ---------------------------------------------------------------------------
__global__ __launch_bounds__(256) void encp_wmma_kernel(
    const float* __restrict__ A,        // (B*T=4096) x 512, f32
    const _Float16* __restrict__ Bp,    // packed We fragments (16 kt x 32 nt)
    float* __restrict__ C) {            // 4096 x 512, f32
  int tid = threadIdx.x;
  int w = tid >> 5, L = tid & 31;
  int half = (L >> 4) & 1;
  int m0 = blockIdx.x * 32 + (w & 1) * 16;
  int g = w >> 1;                       // col group 0..3
  const float* arow = A + (size_t)(m0 + (L & 15)) * ENC_;

  v8f acc[8];
#pragma unroll
  for (int n = 0; n < 8; ++n) acc[n] = (v8f){0.f,0.f,0.f,0.f,0.f,0.f,0.f,0.f};

  for (int kt = 0; kt < 16; ++kt) {
    int kb = kt * 32 + half * 8;
    AFrag a;
#pragma unroll
    for (int j = 0; j < 4; ++j) {
      v2f e0 = *(const v2f*)(arow + kb + 2 * j);
      v2f e1 = *(const v2f*)(arow + kb + 16 + 2 * j);
      a.p[j]     = pk2(e0.x, e0.y);
      a.p[4 + j] = pk2(e1.x, e1.y);
    }
#pragma unroll
    for (int n = 0; n < 8; ++n) {
      int f = kt * 32 + g * 8 + n;
      v16h bf = *(const v16h*)(Bp + (size_t)f * 512 + (size_t)L * 16);
      acc[n] = __builtin_amdgcn_wmma_f32_16x16x32_f16(
          false, a.v, false, bf, (short)0, acc[n], false, false);
    }
  }

  int nnb = g * 128 + (L & 15);
#pragma unroll
  for (int n = 0; n < 8; ++n)
#pragma unroll
    for (int j = 0; j < 8; ++j) {
      int M = j + half * 8;
      C[(size_t)(m0 + M) * JOIN_ + nnb + n * 16] = acc[n][j];
    }
}

// ---------------------------------------------------------------------------
// Fused joint:  logits[b,t,u,:] = mask * (relu(encp[b,t,:] + predp[b,u,:]) @ Wj2 + bj2)
//
// - M = flattened (t,u): 33280 rows per b; block = 256 rows (130 blocks/b).
// - Packed Wj2 (128 KB f16) is staged once per block into LDS with
//   GLOBAL_LOAD_ASYNC_TO_LDS_B128 (ASYNCcnt), then read with ds_load_b128.
// - Each wave computes 2 M-tiles (32 rows) x 128 cols: every B fragment read
//   from LDS feeds two independent WMMAs (latency hiding + half the B traffic).
// - h never materialized: A fragments built in registers
//   (v2f loads from encp/predp -> add -> relu -> v_cvt_pk_rtz).
// ---------------------------------------------------------------------------
__global__ __launch_bounds__(256) void joint_wmma_kernel(
    const float* __restrict__ encp,     // B x T x 512
    const float* __restrict__ predp,    // B x 65 x 512 (bj1 folded in)
    const _Float16* __restrict__ Bp,    // packed Wj2 fragments (16 kt x 8 nt)
    const float* __restrict__ bj2,      // 128
    const int* __restrict__ enc_sz, const int* __restrict__ tgt_sz,
    float* __restrict__ out) {          // B x T x 65 x 128
  __shared__ _Float16 sB[V_ * JOIN_];   // 65536 halfs = 128 KB (WGP LDS is 320 KB)

  const int BLOCKS_PER_B = ROWS_PER_B / 256;          // 130
  int bidx = blockIdx.x;
  int b = bidx / BLOCKS_PER_B;
  int mblk = (bidx - b * BLOCKS_PER_B) * 256;
  int tid = threadIdx.x;
  int w = tid >> 5, L = tid & 31;
  int half = (L >> 4) & 1;
  int m0 = mblk + w * 32;               // this wave's 2 tiles: rows m0..m0+31

  // --- Stage packed Wj2 into LDS: 131072 B = 256 threads * 32 passes * 16 B,
  //     via the CDNA5 async-to-LDS path (no VGPR round trip, ASYNCcnt).
  {
    unsigned lbase = (unsigned)(size_t)(&sB[0]);
    const char* gsrc = (const char*)Bp;
#pragma unroll 4
    for (int p = 0; p < 32; ++p) {
      unsigned off = (unsigned)tid * 16u + (unsigned)p * 4096u;
      unsigned laddr = lbase + off;
      const char* g = gsrc + off;
      asm volatile("global_load_async_to_lds_b128 %0, %1, off"
                   :: "v"(laddr), "v"(g) : "memory");
    }
    asm volatile("s_wait_asynccnt 0x0" ::: "memory");
    __syncthreads();
  }

  // --- Per-tile A-row coordinates for this lane
  const float* erow[2];
  const float* prow[2];
#pragma unroll
  for (int i = 0; i < 2; ++i) {
    unsigned r = (unsigned)(m0 + i * 16 + (L & 15));
    unsigned t = r / 65u;
    unsigned u = r - t * 65u;
    erow[i] = encp  + ((size_t)b * T_   + t) * JOIN_;
    prow[i] = predp + ((size_t)b * UP1_ + u) * JOIN_;
  }
  int es = enc_sz[b], ts = tgt_sz[b];

  v8f acc[2][8];
#pragma unroll
  for (int i = 0; i < 2; ++i)
#pragma unroll
    for (int n = 0; n < 8; ++n)
      acc[i][n] = (v8f){0.f,0.f,0.f,0.f,0.f,0.f,0.f,0.f};

  for (int kt = 0; kt < 16; ++kt) {
    int kb = kt * 32 + half * 8;
    AFrag a[2];
#pragma unroll
    for (int i = 0; i < 2; ++i)
#pragma unroll
      for (int j = 0; j < 4; ++j) {
        v2f e0 = *(const v2f*)(erow[i] + kb + 2 * j);
        v2f p0 = *(const v2f*)(prow[i] + kb + 2 * j);
        v2f e1 = *(const v2f*)(erow[i] + kb + 16 + 2 * j);
        v2f p1 = *(const v2f*)(prow[i] + kb + 16 + 2 * j);
        a[i].p[j]     = pk2(fmaxf(e0.x + p0.x, 0.f), fmaxf(e0.y + p0.y, 0.f));
        a[i].p[4 + j] = pk2(fmaxf(e1.x + p1.x, 0.f), fmaxf(e1.y + p1.y, 0.f));
      }
#pragma unroll
    for (int n = 0; n < 8; ++n) {
      const _Float16* bp = sB + (size_t)(kt * 8 + n) * 512 + (size_t)L * 16;
      v16h bf = *(const v16h*)bp;       // ds_load_b128 x2
      acc[0][n] = __builtin_amdgcn_wmma_f32_16x16x32_f16(
          false, a[0].v, false, bf, (short)0, acc[0][n], false, false);
      acc[1][n] = __builtin_amdgcn_wmma_f32_16x16x32_f16(
          false, a[1].v, false, bf, (short)0, acc[1][n], false, false);
    }
  }

  // --- Epilogue: +bj2, mask (store 0 when masked so full d_out is written)
  int nlane = L & 15;
  float bias[8];
#pragma unroll
  for (int n = 0; n < 8; ++n) bias[n] = bj2[nlane + n * 16];

#pragma unroll
  for (int i = 0; i < 2; ++i)
#pragma unroll
    for (int j = 0; j < 8; ++j) {
      int M = j + half * 8;
      unsigned rr = (unsigned)(m0 + i * 16 + M);
      unsigned tt = rr / 65u;
      unsigned uu = rr - tt * 65u;
      bool valid = ((int)tt < es) && ((int)uu <= ts);
      size_t base = (((size_t)b * T_ + tt) * UP1_ + uu) * V_ + nlane;
#pragma unroll
      for (int n = 0; n < 8; ++n) {
        float v = acc[i][n][j] + bias[n];
        out[base + n * 16] = valid ? v : 0.0f;
      }
    }
}

// ---------------------------------------------------------------------------
// Host side
// ---------------------------------------------------------------------------
extern "C" void kernel_launch(void* const* d_in, const int* in_sizes, int n_in,
                              void* d_out, int out_size, void* d_ws, size_t ws_size,
                              hipStream_t stream) {
  (void)in_sizes; (void)n_in; (void)out_size; (void)ws_size;
  const float* enc     = (const float*)d_in[0];
  const int*   enc_sz  = (const int*)  d_in[1];
  const int*   targets = (const int*)  d_in[2];
  const int*   tgt_sz  = (const int*)  d_in[3];
  const float* emb     = (const float*)d_in[4];
  const float* W1      = (const float*)d_in[5];
  const float* b1      = (const float*)d_in[6];
  const float* W2      = (const float*)d_in[7];
  const float* b2      = (const float*)d_in[8];
  const float* Wj1     = (const float*)d_in[9];
  const float* bj1     = (const float*)d_in[10];
  const float* Wj2     = (const float*)d_in[11];
  const float* bj2     = (const float*)d_in[12];

  // Workspace layout (all offsets 256B-aligned)
  char* ws = (char*)d_ws;
  float*    pred    = (float*)   (ws + 0);          //  520*256*4   = 532480
  float*    predp   = (float*)   (ws + 532480);     //  520*512*4   = 1064960
  float*    encp    = (float*)   (ws + 1597440);    // 4096*512*4   = 8388608
  _Float16* wepack  = (_Float16*)(ws + 9986048);    //  512 frags   = 524288
  _Float16* wj2pack = (_Float16*)(ws + 10510336);   //  128 frags   = 131072
                                                    // total ~10.6 MB

  // 1. Pack weights into WMMA B-fragment order (f16)
  pack_b_kernel<<<64, 256, 0, stream>>>(Wj1, wepack, JOIN_, 32);   // We: 16x32 frags
  pack_b_kernel<<<16, 256, 0, stream>>>(Wj2, wj2pack, V_, 8);      // Wj2: 16x8 frags

  // 2. Prediction network (tiny)
  pred_mlp_kernel<<<B_ * UP1_, 256, 0, stream>>>(targets, emb, W1, b1, W2, b2, pred);

  // 3. pred_proj = pred @ Wp + bj1 (tiny)
  predp_kernel<<<dim3(B_ * UP1_, 2), 256, 0, stream>>>(pred, Wj1, bj1, predp);

  // 4. enc_proj = enc @ We (WMMA)
  encp_wmma_kernel<<<(B_ * T_) / 32, 256, 0, stream>>>(enc, wepack, encp);

  // 5. Fused joint: async LDS staging + relu-add broadcast + 34.9 GFLOP WMMA GEMM
  joint_wmma_kernel<<<B_ * (ROWS_PER_B / 256), 256, 0, stream>>>(
      encp, predp, wj2pack, bj2, enc_sz, tgt_sz, (float*)d_out);
}
